// LINDAAgent_50483045597756
// MI455X (gfx1250) — matrix operand order repaired and checked by reading, not compiled, not run on hardware
//
#include <hip/hip_runtime.h>
#include <hip/hip_bf16.h>
#include <math.h>

// ---------------------------------------------------------------------------
// LINDA agent forward on MI455X (gfx1250, wave32, WMMA f32_16x16x32_f16)
//
// Sizes: ROWS = 8192*8 = 65536, INP=96, RNN=HID=64, ENC=3, NAG=8, NACT=14
// Outputs (d_out, fp32): x_out[65536*14] | h[65536*64] | KL[8192]
//
// Pairwise head is decomposed: H@po1 = U[b,l] + V[b,k] with
// U = h@po1[:64], V = h@po1[64:]; BN stats of U+V derived from per-b sums:
//   sum  += 8*(Su+Sv),  sumsq += 8*(Qu+Qv) + 2*Su*Sv
// Bias columns feeding a batch-norm (aw1_b, po1_b) cancel and are dropped.
// ---------------------------------------------------------------------------

typedef __attribute__((ext_vector_type(16))) _Float16 v16h;
typedef __attribute__((ext_vector_type(8)))  float    v8f;

#define NROWS   65536l
#define NBATCH  8192l
#define INP     96
#define RNNW    64
#define NAGW    8
#define CLIPV   0.2f
#define BNEPS   1e-5f

__device__ __forceinline__ v8f wmma16(v16h a, v16h b, v8f c) {
  // v_wmma_f32_16x16x32_f16, f32 accumulate
  return __builtin_amdgcn_wmma_f32_16x16x32_f16(false, a, false, b, (short)0, c,
                                                false, false);
}

// A-fragment (16x32 f16) from f16 LDS, row-major [16][ldk].
// ISA layout: lane(0-15)=row M; VGPR j holds K = (j>>2)*16 + (lane>>4)*8 + (j&3)*2 .. +1
__device__ __forceinline__ v16h fragA_lds(const _Float16* p0, int ldk, int kbase, int lane) {
  union { v16h h; unsigned u[8]; } f;
  const int r = lane & 15, kh = (lane >> 4) * 8;
  const _Float16* p = p0 + r * ldk + kbase;
#pragma unroll
  for (int j = 0; j < 8; ++j) {
    int kk = ((j >> 2) << 4) + kh + ((j & 3) << 1);
    f.u[j] = *(const unsigned*)(p + kk);       // packed pair (kk, kk+1)
  }
  return f.h;
}

// A-fragment converted on the fly from global fp32 rows.
__device__ __forceinline__ v16h fragA_gf32(const float* p0, long ldk, int kbase, int lane) {
  v16h f;
  const int r = lane & 15, kh = (lane >> 4) * 8;
  const float* p = p0 + (long)r * ldk + kbase;
#pragma unroll
  for (int j = 0; j < 8; ++j) {
    int kk = ((j >> 2) << 4) + kh + ((j & 3) << 1);
    f[2 * j]     = (_Float16)p[kk];
    f[2 * j + 1] = (_Float16)p[kk + 1];
  }
  return f;
}

// B-fragment (32x16 f16) from transposed weights in LDS: [Ntot][Ktot] (= B^T).
// lane(0-15)=col N; VGPR j holds K = (j>>2)*16 + (lane>>4)*8 + (j&3)*2 .. +1
// (same K order as the A-fragment).
__device__ __forceinline__ v16h fragB_ldsT(const _Float16* p0, int ldk, int nbase,
                                           int kbase, int lane) {
  union { v16h h; unsigned u[8]; } f;
  const int n = lane & 15, kh = (lane >> 4) * 8;
  const _Float16* p = p0 + (nbase + n) * ldk + kbase;
#pragma unroll
  for (int j = 0; j < 8; ++j) f.u[j] = *(const unsigned*)(p + ((j >> 2) << 4) + kh + ((j & 3) << 1));
  return f.h;
}

// ---------------------------------------------------------------------------
__global__ void kZero(float* __restrict__ g) { g[threadIdx.x] = 0.f; }

// ---------------------------------------------------------------------------
// Kernel A: fc1 + GRU + {t1, U, V} + BN statistics. 512 blocks x 256 thr;
// each wave owns a 16-row tile (block = 128 rows). 84 WMMAs / wave.
// ---------------------------------------------------------------------------
__global__ __launch_bounds__(256) void kA(
    const float* __restrict__ inputs, const float* __restrict__ h_in,
    const float* __restrict__ fc1_w, const float* __restrict__ fc1_b,
    const float* __restrict__ w_ih,  const float* __restrict__ w_hh,
    const float* __restrict__ b_ih,  const float* __restrict__ b_hh,
    const float* __restrict__ aw1_w, const float* __restrict__ po1_w,
    float* __restrict__ h_out, float* __restrict__ t1_ws,
    float* __restrict__ U_ws,  float* __restrict__ V_ws,
    float* __restrict__ gstats)
{
  __shared__ __align__(16) _Float16 sFC1[64 * 96];     // fc1_w^T  [n][k]
  __shared__ __align__(16) _Float16 sIH [192 * 64];    // w_ih^T
  __shared__ __align__(16) _Float16 sHH [192 * 64];    // w_hh^T
  __shared__ __align__(16) _Float16 sAW1[64 * 64];     // aw1_w^T
  __shared__ __align__(16) _Float16 sPOA[64 * 64];     // po1_w[:64]^T
  __shared__ __align__(16) _Float16 sPOB[64 * 64];     // po1_w[64:]^T
  __shared__ __align__(16) _Float16 sStage[8][16 * 64];// per-wave x/h staging
  __shared__ float sStat[256];                         // bn1 s|q, bn2 s|q

  const int tid = threadIdx.x;
  for (int i = tid; i < 64 * 96; i += 256) {
    int n = i / 96, k = i % 96; sFC1[i] = (_Float16)fc1_w[k * 64 + n];
  }
  for (int i = tid; i < 192 * 64; i += 256) {
    int n = i / 64, k = i % 64;
    sIH[i] = (_Float16)w_ih[k * 192 + n];
    sHH[i] = (_Float16)w_hh[k * 192 + n];
  }
  for (int i = tid; i < 64 * 64; i += 256) {
    int n = i / 64, k = i % 64;
    sAW1[i] = (_Float16)aw1_w[k * 64 + n];
    sPOA[i] = (_Float16)po1_w[k * 64 + n];
    sPOB[i] = (_Float16)po1_w[(64 + k) * 64 + n];
  }
  sStat[tid] = 0.f;
  __syncthreads();

  const int wave = tid >> 5, lane = tid & 31;
  const long R0   = (long)blockIdx.x * 128 + wave * 16;
  const int  n    = lane & 15;
  const int  rowc = (lane >> 4) * 8;                   // C-frag row base
  _Float16* stage = sStage[wave];

  // ---- x = relu(in @ fc1 + b) -> staged f16 ----
  v16h inA[3];
#pragma unroll
  for (int kc = 0; kc < 3; ++kc) inA[kc] = fragA_gf32(inputs + R0 * INP, INP, kc * 32, lane);
#pragma unroll
  for (int t = 0; t < 4; ++t) {
    v8f acc = {};
#pragma unroll
    for (int kc = 0; kc < 3; ++kc)
      acc = wmma16(inA[kc], fragB_ldsT(sFC1, 96, t * 16, kc * 32, lane), acc);
    float bb = fc1_b[t * 16 + n];
#pragma unroll
    for (int i = 0; i < 8; ++i) {
      float v = acc[i] + bb; v = v > 0.f ? v : 0.f;
      stage[(rowc + i) * 64 + t * 16 + n] = (_Float16)v;
    }
  }
  __syncthreads();

  v16h xA[2] = { fragA_lds(stage, 64, 0, lane),  fragA_lds(stage, 64, 32, lane) };
  v16h hA[2] = { fragA_gf32(h_in + R0 * 64, 64, 0, lane),
                 fragA_gf32(h_in + R0 * 64, 64, 32, lane) };

  // ---- GRU: gates per 16-column group ----
#pragma unroll
  for (int t = 0; t < 4; ++t) {
    v8f gr = {}, gz = {}, gn = {}, hr = {}, hz = {}, hn = {};
#pragma unroll
    for (int kc = 0; kc < 2; ++kc) {
      gr = wmma16(xA[kc], fragB_ldsT(sIH, 64,        t * 16, kc * 32, lane), gr);
      gz = wmma16(xA[kc], fragB_ldsT(sIH, 64,  64  + t * 16, kc * 32, lane), gz);
      gn = wmma16(xA[kc], fragB_ldsT(sIH, 64,  128 + t * 16, kc * 32, lane), gn);
      hr = wmma16(hA[kc], fragB_ldsT(sHH, 64,        t * 16, kc * 32, lane), hr);
      hz = wmma16(hA[kc], fragB_ldsT(sHH, 64,  64  + t * 16, kc * 32, lane), hz);
      hn = wmma16(hA[kc], fragB_ldsT(sHH, 64,  128 + t * 16, kc * 32, lane), hn);
    }
    const int cr = t * 16 + n;
    const float bir = b_ih[cr], biz = b_ih[64 + cr], bin_ = b_ih[128 + cr];
    const float bhr = b_hh[cr], bhz = b_hh[64 + cr], bhn  = b_hh[128 + cr];
#pragma unroll
    for (int i = 0; i < 8; ++i) {
      float hv = h_in[(R0 + rowc + i) * 64 + cr];
      float r  = 1.f / (1.f + __expf(-(gr[i] + bir + hr[i] + bhr)));
      float z  = 1.f / (1.f + __expf(-(gz[i] + biz + hz[i] + bhz)));
      float nt = tanhf(gn[i] + bin_ + r * (hn[i] + bhn));
      float h  = (1.f - z) * nt + z * hv;
      h_out[(R0 + rowc + i) * 64 + cr] = h;
      stage[(rowc + i) * 64 + cr]      = (_Float16)h;
    }
  }
  __syncthreads();

  v16h h2A[2] = { fragA_lds(stage, 64, 0, lane), fragA_lds(stage, 64, 32, lane) };

  // ---- t1 = h@aw1, U = h@po1[:64], V = h@po1[64:]  (+ BN statistics) ----
#pragma unroll
  for (int t = 0; t < 4; ++t) {
    v8f a1 = {}, au = {}, av = {};
#pragma unroll
    for (int kc = 0; kc < 2; ++kc) {
      a1 = wmma16(h2A[kc], fragB_ldsT(sAW1, 64, t * 16, kc * 32, lane), a1);
      au = wmma16(h2A[kc], fragB_ldsT(sPOA, 64, t * 16, kc * 32, lane), au);
      av = wmma16(h2A[kc], fragB_ldsT(sPOB, 64, t * 16, kc * 32, lane), av);
    }
    const int c = t * 16 + n;
    float s1 = 0.f, q1 = 0.f, su = 0.f, qu = 0.f, sv = 0.f, qv = 0.f;
#pragma unroll
    for (int i = 0; i < 8; ++i) {         // this lane's 8 rows == one batch b's 8 agents
      long row = R0 + rowc + i;
      float x1 = a1[i], xu = au[i], xv = av[i];
      t1_ws[row * 64 + c] = x1;
      U_ws [row * 64 + c] = xu;
      V_ws [row * 64 + c] = xv;
      s1 += x1; q1 += x1 * x1; su += xu; qu += xu * xu; sv += xv; qv += xv * xv;
    }
    atomicAdd(&sStat[c],        s1);
    atomicAdd(&sStat[64  + c],  q1);
    atomicAdd(&sStat[128 + c],  8.f * (su + sv));                 // pairwise sum
    atomicAdd(&sStat[192 + c],  8.f * (qu + qv) + 2.f * su * sv); // pairwise sumsq
  }
  __syncthreads();
  atomicAdd(&gstats[tid], sStat[tid]);
}

// ---------------------------------------------------------------------------
// Kernel B: BN1 + aw2 + reparam (c) + fc2 -> x_out. 512 blocks x 256 thr.
// ---------------------------------------------------------------------------
__global__ __launch_bounds__(256) void kB(
    const float* __restrict__ noise,
    const float* __restrict__ aw_g,  const float* __restrict__ aw_be,
    const float* __restrict__ aw2_w, const float* __restrict__ aw2_b,
    const float* __restrict__ fc2_w, const float* __restrict__ fc2_b,
    const float* __restrict__ h_full, const float* __restrict__ t1_ws,
    const float* __restrict__ gstats,
    float* __restrict__ xout, float* __restrict__ mu_ws, float* __restrict__ sig_ws)
{
  __shared__ __align__(16) _Float16 sAW2[48 * 64];       // aw2^T
  __shared__ __align__(16) _Float16 sFC2[16 * 96];       // fc2^T padded 14->16, 88->96
  __shared__ __align__(16) _Float16 sStage[8][16 * 96];  // per-wave [h|c|pad] f16
  __shared__ float sMu[8][16 * 24];
  __shared__ float sSg[8][16 * 24];
  __shared__ float sA1[64], sB1[64];

  const int tid = threadIdx.x;
  for (int i = tid; i < 48 * 64; i += 256) {
    int nn = i / 64, k = i % 64; sAW2[i] = (_Float16)aw2_w[k * 48 + nn];
  }
  for (int i = tid; i < 16 * 96; i += 256) {
    int nn = i / 96, k = i % 96;
    sFC2[i] = (nn < 14 && k < 88) ? (_Float16)fc2_w[k * 14 + nn] : (_Float16)0.f;
  }
  if (tid < 64) {                          // BN1 affine: a = t*A1 + B1
    float m   = gstats[tid] * (1.f / 65536.f);
    float var = gstats[64 + tid] * (1.f / 65536.f) - m * m;
    float a   = aw_g[tid] * rsqrtf(var + BNEPS);
    sA1[tid] = a; sB1[tid] = aw_be[tid] - a * m;
  }
  __syncthreads();

  const int wave = tid >> 5, lane = tid & 31;
  const long R0   = (long)blockIdx.x * 128 + wave * 16;
  const int  n    = lane & 15, rowc = (lane >> 4) * 8;
  _Float16* stage = sStage[wave];
  float* muL = sMu[wave];
  float* sgL = sSg[wave];

  // a = relu(bn(t1)) directly into A-fragments (BN is per-K here)
  v16h aA[2];
  {
    const int r = lane & 15, kh = (lane >> 4) * 8;
    const float* p = t1_ws + (R0 + r) * 64;
#pragma unroll
    for (int kc = 0; kc < 2; ++kc) {
      v16h f;
#pragma unroll
      for (int j = 0; j < 8; ++j) {
        int kk = kc * 32 + ((j >> 2) << 4) + kh + ((j & 3) << 1);
        float v0 = p[kk]     * sA1[kk]     + sB1[kk];     v0 = v0 > 0.f ? v0 : 0.f;
        float v1 = p[kk + 1] * sA1[kk + 1] + sB1[kk + 1]; v1 = v1 > 0.f ? v1 : 0.f;
        f[2 * j] = (_Float16)v0; f[2 * j + 1] = (_Float16)v1;
      }
      aA[kc] = f;
    }
  }

  // aw = a @ aw2 + b  -> mu | sigma
#pragma unroll
  for (int t = 0; t < 3; ++t) {
    v8f acc = {};
    acc = wmma16(aA[0], fragB_ldsT(sAW2, 64, t * 16, 0,  lane), acc);
    acc = wmma16(aA[1], fragB_ldsT(sAW2, 64, t * 16, 32, lane), acc);
    const int c = t * 16 + n;
    const float bb = aw2_b[c];
#pragma unroll
    for (int i = 0; i < 8; ++i) {
      float v = acc[i] + bb;
      if (c < 24) muL[(rowc + i) * 24 + c]        = v;
      else        sgL[(rowc + i) * 24 + (c - 24)] = fmaxf(__expf(v), CLIPV);
    }
  }
  __syncthreads();

  // store mu/sigma, build c and [h|c|0] staging
#pragma unroll
  for (int q = 0; q < 12; ++q) {
    int idx = lane * 12 + q;                  // 0..383 = 16 rows x 24
    int row = idx / 24, m = idx % 24;
    long grow = R0 + row;
    float mu = muL[idx], sg = sgL[idx];
    mu_ws [grow * 24 + m] = mu;
    sig_ws[grow * 24 + m] = sg;
    float cc = mu + sqrtf(sg) * noise[grow * 24 + m];
    stage[row * 96 + 64 + m] = (_Float16)cc;
  }
#pragma unroll
  for (int q = 0; q < 32; ++q) {
    int idx = lane * 32 + q; int row = idx >> 6, col = idx & 63;
    stage[row * 96 + col] = (_Float16)h_full[(R0 + row) * 64 + col];
  }
#pragma unroll
  for (int q = 0; q < 4; ++q) {
    int idx = lane * 4 + q; int row = idx >> 3, col = 88 + (idx & 7);
    stage[row * 96 + col] = (_Float16)0.f;
  }
  __syncthreads();

  // x_out = [h|c] @ fc2 + b   (K padded to 96, N padded to 16)
  v8f xo = {};
#pragma unroll
  for (int kc = 0; kc < 3; ++kc)
    xo = wmma16(fragA_lds(stage, 96, kc * 32, lane),
                fragB_ldsT(sFC2, 96, 0, kc * 32, lane), xo);
  if (n < 14) {
    const float bb = fc2_b[n];
#pragma unroll
    for (int i = 0; i < 8; ++i) xout[(R0 + rowc + i) * 14 + n] = xo[i] + bb;
  }
}

// ---------------------------------------------------------------------------
// Kernel C: pairwise head + KL, one block per batch element b (8192 blocks).
// t2[b,k,l] = U[b,l] + V[b,k]; p = relu(t2*A2 + B2); out = p@po2 + b; KL.
// ---------------------------------------------------------------------------
__global__ __launch_bounds__(256) void kC(
    const float* __restrict__ po_g,  const float* __restrict__ po_be,
    const float* __restrict__ po2_w, const float* __restrict__ po2_b,
    const float* __restrict__ U_ws,  const float* __restrict__ V_ws,
    const float* __restrict__ mu_ws, const float* __restrict__ sig_ws,
    const float* __restrict__ gstats,
    float* __restrict__ KL)
{
  __shared__ float sU[8 * 64], sV[8 * 64];
  __shared__ float sP[64 * 64];
  __shared__ float sOut[64 * 6];
  __shared__ float sA2[64], sB2[64], sW2[64 * 6];
  __shared__ float sRed[256];

  const int tid = threadIdx.x;
  const long b = blockIdx.x;
  if (tid < 64) {                          // BN2 affine (po1_b cancels)
    float m   = gstats[128 + tid] * (1.f / 524288.f);
    float var = gstats[192 + tid] * (1.f / 524288.f) - m * m;
    float a   = po_g[tid] * rsqrtf(var + BNEPS);
    sA2[tid] = a; sB2[tid] = po_be[tid] - a * m;
  }
  for (int i = tid; i < 64 * 6; i += 256) sW2[i] = po2_w[i];
  for (int i = tid; i < 512;    i += 256) { sU[i] = U_ws[b * 512 + i]; sV[i] = V_ws[b * 512 + i]; }
  __syncthreads();

  for (int i = tid; i < 4096; i += 256) {          // p rows: pr = k*8 + l
    int pr = i >> 6, c = i & 63, k = pr >> 3, l = pr & 7;
    float v = (sU[l * 64 + c] + sV[k * 64 + c]) * sA2[c] + sB2[c];
    sP[i] = v > 0.f ? v : 0.f;
  }
  __syncthreads();

  for (int i = tid; i < 384; i += 256) {           // out = p @ po2 + b
    int pr = i / 6, e = i % 6;
    float acc = po2_b[e];
    const float* pp = sP + pr * 64;
#pragma unroll
    for (int c = 0; c < 64; ++c) acc += pp[c] * sW2[c * 6 + e];
    sOut[i] = acc;
  }
  __syncthreads();

  float kl = 0.f;
  if (tid < 192) {                                 // 8 agents x 24 channels
    int k = tid / 24, m = tid % 24, lq = m / 6, e = m % 6;
    float mu1 = sOut[(k * 8 + lq) * 6 + e];
    float s1  = fmaxf(__expf(sOut[(k * 8 + 4 + lq) * 6 + e]), CLIPV);
    long rr = b * 8 + k;
    float mu = mu_ws[rr * 24 + m], sg = sig_ws[rr * 24 + m];
    float d = mu - mu1;
    kl = 0.5f * (__logf(s1 / sg) + (sg + d * d) / s1 - 1.f);
  }
  sRed[tid] = kl;
  __syncthreads();
  for (int s = 128; s > 0; s >>= 1) {
    if (tid < s) sRed[tid] += sRed[tid + s];
    __syncthreads();
  }
  if (tid == 0) KL[b] = sRed[0] * (1.f / 192.f);
}

// ---------------------------------------------------------------------------
extern "C" void kernel_launch(void* const* d_in, const int* in_sizes, int n_in,
                              void* d_out, int out_size, void* d_ws, size_t ws_size,
                              hipStream_t stream) {
  (void)in_sizes; (void)n_in; (void)out_size; (void)ws_size;

  const float* inputs = (const float*)d_in[0];
  const float* hidden = (const float*)d_in[1];
  const float* noise  = (const float*)d_in[2];
  const float* fc1_w  = (const float*)d_in[3];
  const float* fc1_b  = (const float*)d_in[4];
  const float* w_ih   = (const float*)d_in[5];
  const float* w_hh   = (const float*)d_in[6];
  const float* b_ih   = (const float*)d_in[7];
  const float* b_hh   = (const float*)d_in[8];
  const float* aw1_w  = (const float*)d_in[9];
  /* aw1_b = d_in[10]: cancels inside batch-norm */
  const float* aw_g   = (const float*)d_in[11];
  const float* aw_be  = (const float*)d_in[12];
  const float* aw2_w  = (const float*)d_in[13];
  const float* aw2_b  = (const float*)d_in[14];
  const float* po1_w  = (const float*)d_in[15];
  /* po1_b = d_in[16]: cancels inside batch-norm */
  const float* po_g   = (const float*)d_in[17];
  const float* po_be  = (const float*)d_in[18];
  const float* po2_w  = (const float*)d_in[19];
  const float* po2_b  = (const float*)d_in[20];
  const float* fc2_w  = (const float*)d_in[21];
  const float* fc2_b  = (const float*)d_in[22];

  float* out   = (float*)d_out;
  float* xout  = out;                          // 65536*14
  float* hout  = out  + NROWS * 14;            // 65536*64
  float* klout = hout + NROWS * 64;            // 8192

  float* ws    = (float*)d_ws;                 // needs ~63 MB
  float* t1    = ws;                           // 65536*64
  float* U     = t1  + NROWS * 64;             // 65536*64
  float* V     = U   + NROWS * 64;             // 65536*64
  float* muw   = V   + NROWS * 64;             // 65536*24
  float* sgw   = muw + NROWS * 24;             // 65536*24
  float* stats = sgw + NROWS * 24;             // 256 (bn1 s|q, bn2 s|q)

  kZero<<<1, 256, 0, stream>>>(stats);
  kA<<<512, 256, 0, stream>>>(inputs, hidden, fc1_w, fc1_b, w_ih, w_hh, b_ih, b_hh,
                              aw1_w, po1_w, hout, t1, U, V, stats);
  kB<<<512, 256, 0, stream>>>(noise, aw_g, aw_be, aw2_w, aw2_b, fc2_w, fc2_b,
                              hout, t1, stats, xout, muw, sgw);
  kC<<<8192, 256, 0, stream>>>(po_g, po_be, po2_w, po2_b, U, V, muw, sgw, stats, klout);
}
